// PointNetExtractor_78280073937389
// MI455X (gfx1250) — matrix operand order repaired and checked by reading, not compiled, not run on hardware
//
#include <hip/hip_runtime.h>
#include <hip/hip_bf16.h>
#include <cstdint>
#include <cstddef>

// ---------------------------------------------------------------------------
// PointNet++ (MSG) feature extractor for MI455X / gfx1250, wave32 + WMMA.
// All matrix work runs through v_wmma_f32_16x16x32_f16 (f16 in, f32 acc).
// Weights are pre-folded (BN) + converted to f16 + K-padded once, so the
// WMMA inner loops are pure b128 loads + wmma. Each wave computes a 16x32
// output macro-tile (two accumulators share one A fragment).
// ---------------------------------------------------------------------------

typedef _Float16 v16h __attribute__((ext_vector_type(16)));
typedef _Float16 v8h  __attribute__((ext_vector_type(8)));
typedef float    v8f  __attribute__((ext_vector_type(8)));

#define EPS_BN 1e-5f

// Pre-processed layer: f16 weights (Nout x Kp, row-major, zero-padded) +
// folded f32 bias. y = relu(x @ W^T + bias).
struct WL { const _Float16* W; const float* bias; };

// ---- CDNA5 WMMA 16x16x32 f16 fragment loaders (wave32, ISA 7.12.2) -------
// A 16x32: lane&15 = row, lane>>4 = K-half. Per lane the fragment is two
// contiguous 8-half runs: [kc+8h .. +7] and [kc+16+8h .. +7] -> 2x b128.
__device__ __forceinline__ v16h load_a16(const _Float16* A, int ldk, int row0,
                                         int kc, int lane) {
  const _Float16* p = A + (size_t)(row0 + (lane & 15)) * ldk + kc + 8 * (lane >> 4);
  v8h lo = *(const v8h*)p;
  v8h hi = *(const v8h*)(p + 16);
  return __builtin_shufflevector(lo, hi, 0,1,2,3,4,5,6,7,8,9,10,11,12,13,14,15);
}

// B 32x16 from W' (Nout x Kp): lane&15 = n, lane>>4 = K-half. Per lane the
// fragment is two contiguous 8-half runs: [kc+8h .. +7] and [kc+16+8h .. +7] -> 2x b128.
__device__ __forceinline__ v16h load_b16(const _Float16* Wt, int ldk, int n0,
                                         int kc, int lane) {
  const _Float16* p = Wt + (size_t)(n0 + (lane & 15)) * ldk + kc + 8 * (lane >> 4);
  v8h lo = *(const v8h*)p;
  v8h hi = *(const v8h*)(p + 16);
  return __builtin_shufflevector(lo, hi, 0,1,2,3,4,5,6,7,8,9,10,11,12,13,14,15);
}

// One MLP layer: Bout[M,Nout] = relu(Ain[M,KP] @ W'^T + bias), f16 in/out.
// M, KP compile-time (M mult of 16, KP mult of 32 -> full unroll); Nout a
// multiple of 32. Each wave owns a 16x32 macro-tile: one A fragment feeds
// two accumulators. C/D layout: lane&15 = n, VGPR v = row v + 8*(lane>>4).
template<int KP, int M>
__device__ __forceinline__ void gemm_layer(const _Float16* Ain, WL L, int Nout,
                                           _Float16* Bout, int wave, int lane) {
  constexpr int TM = M >> 4;
  int TN2 = Nout >> 5;                        // pairs of n-tiles
  for (int t = wave; t < TM * TN2; t += 8) {  // wave-uniform => EXEC all ones
    int tm = t % TM, tn = (t / TM) * 2;
    v8f acc0 = {}, acc1 = {};
#pragma unroll
    for (int kc = 0; kc < KP; kc += 32) {
      v16h a  = load_a16(Ain, KP, tm * 16, kc, lane);
      v16h b0 = load_b16(L.W, KP, tn * 16, kc, lane);
      v16h b1 = load_b16(L.W, KP, tn * 16 + 16, kc, lane);
      acc0 = __builtin_amdgcn_wmma_f32_16x16x32_f16(false, a, false, b0,
                                                    (short)0, acc0, false, false);
      acc1 = __builtin_amdgcn_wmma_f32_16x16x32_f16(false, a, false, b1,
                                                    (short)0, acc1, false, false);
    }
    int n0 = tn * 16 + (lane & 15), h = lane >> 4;
    float bs0 = L.bias[n0], bs1 = L.bias[n0 + 16];
#pragma unroll
    for (int vv = 0; vv < 8; ++vv) {
      size_t r = (size_t)(tm * 16 + vv + 8 * h) * Nout;
      Bout[r + n0]      = (_Float16)fmaxf(acc0[vv] + bs0, 0.f);
      Bout[r + n0 + 16] = (_Float16)fmaxf(acc1[vv] + bs1, 0.f);
    }
  }
}

// ---------------------------------------------------------------------------
// Weight prep: fold BN into W, convert to f16, zero-pad K -> Kp.
// If g == nullptr: plain linear layer (scale 1, bias 0).
// ---------------------------------------------------------------------------
__global__ void prep_layer_kernel(const float* __restrict__ W, const float* __restrict__ g,
                                  const float* __restrict__ b, const float* __restrict__ m,
                                  const float* __restrict__ v, int Nout, int K, int Kp,
                                  _Float16* __restrict__ Wh, float* __restrict__ bias) {
  int e = blockIdx.x * blockDim.x + threadIdx.x;
  if (e < Nout * Kp) {
    int n = e / Kp, k = e - n * Kp;
    float sc = 1.f;
    if (g) sc = g[n] * rsqrtf(v[n] + EPS_BN);
    Wh[e] = (k < K) ? (_Float16)(W[(size_t)n * K + k] * sc) : (_Float16)0.f;
  }
  if (e < Nout) {
    float bb = 0.f;
    if (g) {
      float sc = g[e] * rsqrtf(v[e] + EPS_BN);
      bb = b[e] - m[e] * sc;
    }
    bias[e] = bb;
  }
}

// ---------------------------------------------------------------------------
// Farthest point sampling: one block per batch, dists register-resident.
// ---------------------------------------------------------------------------
template<int BLOCK, int PERT>
__global__ __launch_bounds__(BLOCK) void fps_kernel(const float* __restrict__ xyz,
                                                    int stride, int N, int npoint,
                                                    int* __restrict__ out) {
  constexpr int NW = BLOCK / 32;
  __shared__ float swv[NW];
  __shared__ int   swi[NW];
  __shared__ float cur[3];
  int b = blockIdx.x;
  const float* base = xyz + (size_t)b * N * stride;
  int tid = threadIdx.x, lane = tid & 31, wave = tid >> 5;
  float d[PERT];
#pragma unroll
  for (int i = 0; i < PERT; ++i) d[i] = 1e10f;
  if (tid == 0) {
    out[(size_t)b * npoint] = 0;
    cur[0] = base[0]; cur[1] = base[1]; cur[2] = base[2];
  }
  __syncthreads();
  for (int t = 1; t < npoint; ++t) {
    float px = cur[0], py = cur[1], pz = cur[2];
    float best = -1.f; int bi = 0x7fffffff;
#pragma unroll
    for (int i = 0; i < PERT; ++i) {
      int j = tid + i * BLOCK;
      if (j < N) {
        const float* q = base + (size_t)j * stride;
        float dx = q[0] - px, dy = q[1] - py, dz = q[2] - pz;
        float dd = dx * dx + dy * dy + dz * dz;
        d[i] = fminf(d[i], dd);
        if (d[i] > best) { best = d[i]; bi = j; }
      }
    }
#pragma unroll
    for (int o2 = 16; o2 > 0; o2 >>= 1) {  // wave32 argmax, lower idx wins ties
      float ov = __shfl_xor(best, o2, 32);
      int   oi = __shfl_xor(bi,   o2, 32);
      if (ov > best || (ov == best && oi < bi)) { best = ov; bi = oi; }
    }
    if (lane == 0) { swv[wave] = best; swi[wave] = bi; }
    __syncthreads();
    if (wave == 0) {
      float bv = (lane < NW) ? swv[lane] : -1.f;
      int   bj = (lane < NW) ? swi[lane] : 0x7fffffff;
#pragma unroll
      for (int o2 = 16; o2 > 0; o2 >>= 1) {
        float ov = __shfl_xor(bv, o2, 32);
        int   oi = __shfl_xor(bj, o2, 32);
        if (ov > bv || (ov == bv && oi < bj)) { bv = ov; bj = oi; }
      }
      if (lane == 0) {
        out[(size_t)b * npoint + t] = bj;
        const float* q = base + (size_t)bj * stride;
        cur[0] = q[0]; cur[1] = q[1]; cur[2] = q[2];
      }
    }
    __syncthreads();
  }
}

__global__ void gather3_kernel(const float* __restrict__ xyz, int stride, int N,
                               const int* __restrict__ idx, int npoint,
                               float* __restrict__ out, int total) {
  int i = blockIdx.x * blockDim.x + threadIdx.x;
  if (i >= total) return;
  int b = i / npoint;
  int j = idx[i];
  const float* q = xyz + ((size_t)b * N + j) * stride;
  out[(size_t)i * 3 + 0] = q[0];
  out[(size_t)i * 3 + 1] = q[1];
  out[(size_t)i * 3 + 2] = q[2];
}

// ---------------------------------------------------------------------------
// Fused SA scale: ball-query + gather(LDS,f16) + 3x WMMA MLP + max-pool.
// One block (8 waves) per center. grid = B*S. FT = float (raw pointcloud
// feats) or _Float16 (previous stage output, vector-copy gather).
// ---------------------------------------------------------------------------
template<int KNB, int C0, int C0P, int N1, int N2, int N3, typename FT>
__global__ __launch_bounds__(256) void sa_scale_kernel(
    const float* __restrict__ xyz, int xstride, int N,
    const FT* __restrict__ feats, int fstride,
    const float* __restrict__ ctr, int S, float radius,
    WL w1, WL w2, WL w3,
    _Float16* __restrict__ outF, int ostride, int ooff) {
  constexpr int CA = (C0P > N2) ? C0P : N2;
  constexpr int CB = (N1 > N3) ? N1 : N3;
  __shared__ alignas(32) _Float16 bufA[KNB * CA];
  __shared__ alignas(32) _Float16 bufB[KNB * CB];
  __shared__ int nbr[KNB];

  int tid = threadIdx.x, lane = tid & 31, wave = tid >> 5;
  int bs = blockIdx.x;
  int b = bs / S, s = bs - b * S;
  const float* c3 = ctr + (size_t)bs * 3;

  // --- ball query (wave 0): first KNB indices with d2 < r^2, in order ---
  if (wave == 0) {
    float px = c3[0], py = c3[1], pz = c3[2];
    float r2 = radius * radius;
    const float* bx = xyz + (size_t)b * N * xstride;
    int cnt = 0;
    for (int j0 = 0; j0 < N && cnt < KNB; j0 += 32) {
      int j = j0 + lane;
      bool in = false;
      if (j < N) {
        const float* q = bx + (size_t)j * xstride;
        float dx = q[0] - px, dy = q[1] - py, dz = q[2] - pz;
        in = (dx * dx + dy * dy + dz * dz) < r2;
      }
      unsigned mask = (unsigned)__ballot(in);
      int pos = cnt + __popc(mask & ((1u << lane) - 1u));
      if (in && pos < KNB) nbr[pos] = j;
      cnt += __popc(mask);
    }
    if (cnt > KNB) cnt = KNB;
    int fill = (cnt > 0) ? nbr[0] : 0;
    for (int p = cnt + lane; p < KNB; p += 32) nbr[p] = fill;
  }
  __syncthreads();

  // --- gather neighbor features into LDS f16 (K zero-padded to C0P) ---
  if constexpr (sizeof(FT) == 2 && (C0 % 8) == 0 && C0 == C0P) {
    constexpr int C8 = C0 / 8;           // 16-byte vector copy, no conversion
    for (int e = tid; e < KNB * C8; e += 256) {
      int mrow = e / C8, c8 = e - mrow * C8;
      const v8h* src = (const v8h*)((const _Float16*)feats +
                        ((size_t)b * N + nbr[mrow]) * fstride) + c8;
      ((v8h*)bufA)[(size_t)mrow * C8 + c8] = *src;
    }
  } else {
    for (int e = tid; e < KNB * C0P; e += 256) {
      int mrow = e / C0P, c = e - mrow * C0P;
      float val = 0.f;
      if (c < C0) val = (float)feats[((size_t)b * N + nbr[mrow]) * fstride + c];
      bufA[e] = (_Float16)val;
    }
  }
  __syncthreads();

  gemm_layer<C0P, KNB>(bufA, w1, N1, bufB, wave, lane);
  __syncthreads();
  gemm_layer<N1, KNB>(bufB, w2, N2, bufA, wave, lane);
  __syncthreads();
  gemm_layer<N2, KNB>(bufA, w3, N3, bufB, wave, lane);
  __syncthreads();

  // --- max over the KNB neighbors ---
  for (int c = tid; c < N3; c += 256) {
    float mx = -1e30f;
    for (int m = 0; m < KNB; ++m) mx = fmaxf(mx, (float)bufB[(size_t)m * N3 + c]);
    outF[((size_t)b * S + s) * ostride + ooff + c] = (_Float16)mx;
  }
}

// ---------------------------------------------------------------------------
// SA3: (B,128,640) f16 -> MLP 256,512,1024 -> max over 128 points -> (B,1024).
// Layer 1 streams A from global f16; layer 3 streams the column-max so the
// 128x1024 activation never materializes (two m-tiles share one B fragment).
// One block per batch element.
// ---------------------------------------------------------------------------
__global__ __launch_bounds__(256) void sa3_kernel(const _Float16* __restrict__ F,
                                                  WL L1, WL L2, WL L3,
                                                  _Float16* __restrict__ gout) {
  constexpr int M = 128, C0 = 640, N1 = 256, N2 = 512, N3 = 1024;
  __shared__ alignas(32) _Float16 sA[M * N2];  // layer-2 output (128 KB)
  __shared__ alignas(32) _Float16 sB[M * N1];  // layer-1 output (64 KB)
  int tid = threadIdx.x, lane = tid & 31, wave = tid >> 5;
  int b = blockIdx.x;

  gemm_layer<C0, M>(F + (size_t)b * M * C0, L1, N1, sB, wave, lane);
  __syncthreads();
  gemm_layer<N1, M>(sB, L2, N2, sA, wave, lane);
  __syncthreads();

  for (int tn = wave; tn < N3 / 16; tn += 8) {
    int n = tn * 16 + (lane & 15);
    float bs = L3.bias[n];
    float cm = 0.f;  // relu output >= 0, so 0 is the identity for this max
    for (int tm = 0; tm < M / 16; tm += 2) {  // two m-tiles share B fragments
      v8f acc0 = {}, acc1 = {};
#pragma unroll
      for (int kc = 0; kc < N2; kc += 32) {
        v16h bb = load_b16(L3.W, N2, tn * 16, kc, lane);
        v16h a0 = load_a16(sA, N2, tm * 16, kc, lane);
        v16h a1 = load_a16(sA, N2, tm * 16 + 16, kc, lane);
        acc0 = __builtin_amdgcn_wmma_f32_16x16x32_f16(false, a0, false, bb,
                                                      (short)0, acc0, false, false);
        acc1 = __builtin_amdgcn_wmma_f32_16x16x32_f16(false, a1, false, bb,
                                                      (short)0, acc1, false, false);
      }
#pragma unroll
      for (int vv = 0; vv < 8; ++vv) {
        cm = fmaxf(cm, fmaxf(acc0[vv] + bs, 0.f));
        cm = fmaxf(cm, fmaxf(acc1[vv] + bs, 0.f));
      }
    }
    cm = fmaxf(cm, __shfl_xor(cm, 16, 32));  // merge the two lane halves (rows)
    if (lane < 16) gout[(size_t)b * N3 + n] = (_Float16)cm;
  }
}

// ---------------------------------------------------------------------------
// FC head: (16,1024) f16 -> relu(@W1'+b1) (16,512) -> @W2' (16,256) f32 out.
// M = 16 is exactly one WMMA M-tile. One block; final layer is 8 n-pairs,
// one per wave.
// ---------------------------------------------------------------------------
__global__ __launch_bounds__(256) void fc_kernel(const _Float16* __restrict__ G,
                                                 WL L1, WL L2,
                                                 float* __restrict__ out) {
  constexpr int M = 16, K1 = 1024, N1 = 512, N2 = 256;
  __shared__ alignas(32) _Float16 sg[M * K1];
  __shared__ alignas(32) _Float16 sh[M * N1];
  int tid = threadIdx.x, lane = tid & 31, wave = tid >> 5;
  for (int e = tid; e < M * K1; e += 256) sg[e] = G[e];
  __syncthreads();
  gemm_layer<K1, M>(sg, L1, N1, sh, wave, lane);
  __syncthreads();
  {
    int tn = wave * 2;                      // 8 waves x 2 n-tiles = N2/16
    v8f acc0 = {}, acc1 = {};
#pragma unroll
    for (int kc = 0; kc < N1; kc += 32) {
      v16h a  = load_a16(sh, N1, 0, kc, lane);
      v16h b0 = load_b16(L2.W, N1, tn * 16, kc, lane);
      v16h b1 = load_b16(L2.W, N1, tn * 16 + 16, kc, lane);
      acc0 = __builtin_amdgcn_wmma_f32_16x16x32_f16(false, a, false, b0,
                                                    (short)0, acc0, false, false);
      acc1 = __builtin_amdgcn_wmma_f32_16x16x32_f16(false, a, false, b1,
                                                    (short)0, acc1, false, false);
    }
    int n = tn * 16 + (lane & 15), h = lane >> 4;
#pragma unroll
    for (int vv = 0; vv < 8; ++vv) {        // final layer: no bn/relu
      size_t r = (size_t)(vv + 8 * h) * N2;
      out[r + n]      = acc0[vv];
      out[r + n + 16] = acc1[vv];
    }
  }
}

// ---------------------------------------------------------------------------
// Host launch. Input order = insertion-order depth-first flatten of
// setup_inputs(): pointcloud, sa1(3 mlps x 3 layers x {W,g,b,m,v}),
// sa2(45), sa3(15), fc(w1, g,b,m,v, w2).
// ---------------------------------------------------------------------------
extern "C" void kernel_launch(void* const* d_in, const int* in_sizes, int n_in,
                              void* d_out, int out_size, void* d_ws, size_t ws_size,
                              hipStream_t stream) {
  (void)in_sizes; (void)n_in; (void)out_size; (void)ws_size;
  const float* pc = (const float*)d_in[0];
  auto P = [&](int i) { return (const float*)d_in[i]; };
  auto sa1i = [](int mlp, int layer) { return 1  + (mlp * 3 + layer) * 5; };
  auto sa2i = [](int mlp, int layer) { return 46 + (mlp * 3 + layer) * 5; };
  auto sa3i = [](int layer)          { return 91 + layer * 5; };

  const int B = 16, N = 8192, S1 = 512, S2 = 128;

  char* wsb = (char*)d_ws;
  size_t off = 0;
  auto alloc = [&](size_t bytes) { size_t r = off; off += (bytes + 255) & ~(size_t)255; return r; };
  int*       idx1 = (int*)      (wsb + alloc((size_t)B * S1 * 4));
  float*     nx1  = (float*)    (wsb + alloc((size_t)B * S1 * 3 * 4));
  _Float16*  f1h  = (_Float16*) (wsb + alloc((size_t)B * S1 * 320 * 2));
  int*       idx2 = (int*)      (wsb + alloc((size_t)B * S2 * 4));
  float*     nx2  = (float*)    (wsb + alloc((size_t)B * S2 * 3 * 4));
  _Float16*  f2h  = (_Float16*) (wsb + alloc((size_t)B * S2 * 640 * 2));
  _Float16*  gbh  = (_Float16*) (wsb + alloc((size_t)B * 1024 * 2));

  // --- weight prep: fold BN, f16, K-padded (23 tiny launches) ---
  auto prep = [&](int base, int Nout, int K, bool bn) -> WL {
    int Kp = (K + 31) & ~31;
    _Float16* Wh  = (_Float16*)(wsb + alloc((size_t)Nout * Kp * 2));
    float*    bia = (float*)   (wsb + alloc((size_t)Nout * 4));
    prep_layer_kernel<<<(Nout * Kp + 255) / 256, 256, 0, stream>>>(
        P(base), bn ? P(base + 1) : nullptr, bn ? P(base + 2) : nullptr,
        bn ? P(base + 3) : nullptr, bn ? P(base + 4) : nullptr,
        Nout, K, Kp, Wh, bia);
    WL r; r.W = Wh; r.bias = bia; return r;
  };
  // SA1 mlps: [3,32,32,64], [3,64,64,128], [3,64,96,128]
  WL s1m0l1 = prep(sa1i(0,0),  32,   3, true), s1m0l2 = prep(sa1i(0,1),  32,  32, true), s1m0l3 = prep(sa1i(0,2),  64,  32, true);
  WL s1m1l1 = prep(sa1i(1,0),  64,   3, true), s1m1l2 = prep(sa1i(1,1),  64,  64, true), s1m1l3 = prep(sa1i(1,2), 128,  64, true);
  WL s1m2l1 = prep(sa1i(2,0),  64,   3, true), s1m2l2 = prep(sa1i(2,1),  96,  64, true), s1m2l3 = prep(sa1i(2,2), 128,  96, true);
  // SA2 mlps: [320,64,64,128], [320,128,128,256], [320,128,128,256]
  WL s2m0l1 = prep(sa2i(0,0),  64, 320, true), s2m0l2 = prep(sa2i(0,1),  64,  64, true), s2m0l3 = prep(sa2i(0,2), 128,  64, true);
  WL s2m1l1 = prep(sa2i(1,0), 128, 320, true), s2m1l2 = prep(sa2i(1,1), 128, 128, true), s2m1l3 = prep(sa2i(1,2), 256, 128, true);
  WL s2m2l1 = prep(sa2i(2,0), 128, 320, true), s2m2l2 = prep(sa2i(2,1), 128, 128, true), s2m2l3 = prep(sa2i(2,2), 256, 128, true);
  // SA3: [640,256,512,1024]
  WL s3l1 = prep(sa3i(0), 256, 640, true), s3l2 = prep(sa3i(1), 512, 256, true), s3l3 = prep(sa3i(2), 1024, 512, true);
  // FC: w1 (512,1024) + bn, w2 (256,512) plain
  WL fc1 = prep(106, 512, 1024, true);
  WL fc2 = prep(111, 256,  512, false);

  // ---- SA1 ----
  fps_kernel<1024, 8><<<B, 1024, 0, stream>>>(pc, 6, N, S1, idx1);
  { int tot = B * S1;
    gather3_kernel<<<(tot + 255) / 256, 256, 0, stream>>>(pc, 6, N, idx1, S1, nx1, tot); }

  sa_scale_kernel<16, 3, 32, 32, 32, 64, float><<<B * S1, 256, 0, stream>>>(
      pc, 6, N, pc + 3, 6, nx1, S1, 0.1f, s1m0l1, s1m0l2, s1m0l3, f1h, 320, 0);
  sa_scale_kernel<32, 3, 32, 64, 64, 128, float><<<B * S1, 256, 0, stream>>>(
      pc, 6, N, pc + 3, 6, nx1, S1, 0.2f, s1m1l1, s1m1l2, s1m1l3, f1h, 320, 64);
  sa_scale_kernel<128, 3, 32, 64, 96, 128, float><<<B * S1, 256, 0, stream>>>(
      pc, 6, N, pc + 3, 6, nx1, S1, 0.4f, s1m2l1, s1m2l2, s1m2l3, f1h, 320, 192);

  // ---- SA2 ----
  fps_kernel<512, 1><<<B, 512, 0, stream>>>(nx1, 3, S1, S2, idx2);
  { int tot = B * S2;
    gather3_kernel<<<(tot + 255) / 256, 256, 0, stream>>>(nx1, 3, S1, idx2, S2, nx2, tot); }

  sa_scale_kernel<32, 320, 320, 64, 64, 128, _Float16><<<B * S2, 256, 0, stream>>>(
      nx1, 3, S1, f1h, 320, nx2, S2, 0.2f, s2m0l1, s2m0l2, s2m0l3, f2h, 640, 0);
  sa_scale_kernel<64, 320, 320, 128, 128, 256, _Float16><<<B * S2, 256, 0, stream>>>(
      nx1, 3, S1, f1h, 320, nx2, S2, 0.4f, s2m1l1, s2m1l2, s2m1l3, f2h, 640, 128);
  sa_scale_kernel<128, 320, 320, 128, 128, 256, _Float16><<<B * S2, 256, 0, stream>>>(
      nx1, 3, S1, f1h, 320, nx2, S2, 0.8f, s2m2l1, s2m2l2, s2m2l3, f2h, 640, 384);

  // ---- SA3 (global MLP + max) ----
  sa3_kernel<<<B, 256, 0, stream>>>(f2h, s3l1, s3l2, s3l3, gbh);

  // ---- FC head ----
  fc_kernel<<<1, 256, 0, stream>>>(gbh, fc1, fc2, (float*)d_out);
}
